// CorrespondenceLoss_21749714387505
// MI455X (gfx1250) — matrix operand order repaired and checked by reading, not compiled, not run on hardware
//
#include <hip/hip_runtime.h>

typedef __attribute__((ext_vector_type(16))) __bf16       v16bf;
typedef __attribute__((ext_vector_type(8)))  float        v8f;
typedef __attribute__((ext_vector_type(4)))  unsigned int u32x4;

#define BATCH 8
#define CHN   64
#define HW    4096
#define NPTS  2048

#define ROW_US    72                    // padded LDS row stride: 144 B (bank-conflict-free)
#define PANEL_DESC 64                   // one grid row of descriptors per panel
#define PANEL_US  (PANEL_DESC * ROW_US) // 4608 ushorts = 9216 B per buffer

static __device__ __forceinline__ unsigned short f2bf(float f) {
  unsigned int u = __float_as_uint(f);
  unsigned int r = u + 0x7FFFu + ((u >> 16) & 1u);   // round-to-nearest-even
  return (unsigned short)(r >> 16);
}
static __device__ __forceinline__ float bf2f(unsigned short h) {
  return __uint_as_float(((unsigned int)h) << 16);
}

// ---------------------------------------------------------------------------
// Kernel 1: column-normalize f1/f2, emit bf16 rows transposed to [desc][chan].
// ---------------------------------------------------------------------------
__global__ __launch_bounds__(256) void normalize_k(const float* __restrict__ x1,
                                                   const float* __restrict__ x2,
                                                   unsigned short* __restrict__ dT1,
                                                   unsigned short* __restrict__ dT2) {
  int idx  = blockIdx.x * 256 + threadIdx.x;     // 0 .. 2*B*HW-1
  int side = idx >> 15;
  int rem  = idx & 32767;
  int b    = rem >> 12;
  int m    = rem & 4095;
  const float* src = (side ? x2 : x1) + (size_t)b * CHN * HW + m;
  float s = 0.f;
#pragma unroll 8
  for (int c = 0; c < CHN; ++c) { float v = src[(size_t)c * HW]; s += v * v; }
  float inv = 1.f / fmaxf(sqrtf(s), 1e-12f);
  unsigned int packed[CHN / 2];
#pragma unroll 8
  for (int c = 0; c < CHN; c += 2) {
    unsigned short lo = f2bf(src[(size_t)c * HW] * inv);
    unsigned short hi = f2bf(src[(size_t)(c + 1) * HW] * inv);
    packed[c >> 1] = (unsigned int)lo | ((unsigned int)hi << 16);
  }
  unsigned short* dst = (side ? dT2 : dT1) + (size_t)(b * HW + m) * CHN;
  u32x4* d4 = (u32x4*)dst;
#pragma unroll
  for (int j = 0; j < 8; ++j) {
    u32x4 v = {packed[4 * j], packed[4 * j + 1], packed[4 * j + 2], packed[4 * j + 3]};
    d4[j] = v;
  }
}

// ---------------------------------------------------------------------------
// Kernel 2: positive[n] = 2 - 2 * <d1[:,ids[n]], d2[:,pos2[n]]>
// ---------------------------------------------------------------------------
__global__ __launch_bounds__(256) void positive_k(const unsigned short* __restrict__ dT1,
                                                  const unsigned short* __restrict__ dT2,
                                                  const int* __restrict__ ids,
                                                  const int* __restrict__ pos2,
                                                  float* __restrict__ posv) {
  int idx = blockIdx.x * 256 + threadIdx.x;      // < B*NPTS
  int b = idx >> 11, n = idx & 2047;
  int i1 = ids[b * NPTS + n];
  int pr = pos2[b * 2 * NPTS + n];
  int pc = pos2[b * 2 * NPTS + NPTS + n];
  int i2 = pr * 64 + pc;
  const unsigned short* r1 = dT1 + (size_t)(b * HW + i1) * CHN;
  const unsigned short* r2 = dT2 + (size_t)(b * HW + i2) * CHN;
  float dot = 0.f;
#pragma unroll 8
  for (int c = 0; c < CHN; ++c) dot += bf2f(r1[c]) * bf2f(r2[c]);
  posv[idx] = 2.f - 2.f * dot;
}

// ---------------------------------------------------------------------------
// Async copy of one 64-descriptor panel (8 KB) into LDS, padded 144B rows.
// 256 threads x 2 chunks x 16B = 8 KB; tracked on ASYNCcnt (2 per wave).
// INST_OFFSET is added to BOTH global and LDS addresses, so offset:16 covers
// the second chunk with the same registers.
// ---------------------------------------------------------------------------
static __device__ __forceinline__ void copy_panel(const unsigned short* __restrict__ cols,
                                                  unsigned lds_base, int panel, int buf, int t) {
  int desc = t >> 2;                     // 0..63
  int sub2 = (t & 3) << 1;               // chunk pair: 2*(t%4), +1
  unsigned goff = (unsigned)((panel * PANEL_DESC + desc) * (CHN * 2) + sub2 * 16);
  unsigned loff = lds_base + (unsigned)(buf * (PANEL_US * 2) + desc * (ROW_US * 2) + sub2 * 16);
  asm volatile("global_load_async_to_lds_b128 %0, %1, %2 offset:0\n\t"
               "global_load_async_to_lds_b128 %0, %1, %2 offset:16"
               :: "v"(loff), "v"(goff), "s"(cols)
               : "memory");
}
static __device__ __forceinline__ void wait_async_le0() {
  asm volatile("s_wait_asynccnt 0x0" ::: "memory");
}

// ---------------------------------------------------------------------------
// Kernel 3: masked min over 4096 columns via bf16 WMMA with LDS-staged,
// double-buffered column panels (async global->LDS DMA overlapped with WMMA).
// A sched_barrier pins the 16-ds_load burst ahead of the WMMA chain so the
// dscnt waits stagger (loads pipelined) instead of stalling each WMMA.
// pass 0: rows = d1[ids], cols = d2_all, mask at pos2      -> neg2
// pass 1: rows = d2[pos2], cols = d1_all, mask at ids-pos  -> neg1
// ---------------------------------------------------------------------------
__global__ __launch_bounds__(256) void negmin_k(const unsigned short* __restrict__ dT1,
                                                const unsigned short* __restrict__ dT2,
                                                const int* __restrict__ ids,
                                                const int* __restrict__ pos2,
                                                float* __restrict__ neg1,
                                                float* __restrict__ neg2) {
  __shared__ __align__(16) unsigned short smem[2 * PANEL_US];   // 18 KB, 2 buffers

  const int lane = threadIdx.x & 31;
  const int wave = threadIdx.x >> 5;
  const int hi   = lane >> 4;
  const int lo   = lane & 15;
  const int b    = blockIdx.y;
  const int pass = blockIdx.z;
  const int nbase = (blockIdx.x * 8 + wave) * 16;

  const unsigned short* rows = (pass ? dT2 : dT1) + (size_t)b * HW * CHN;
  const unsigned short* cols = (pass ? dT1 : dT2) + (size_t)b * HW * CHN;
  const unsigned lds_base = (unsigned)(uintptr_t)&smem[0];

  // Gather this lane's A-row descriptor.
  int nA = nbase + lo;
  int rowDesc;
  if (pass == 0) {
    rowDesc = ids[b * NPTS + nA];
  } else {
    int pr = pos2[b * 2 * NPTS + nA];
    int pc = pos2[b * 2 * NPTS + NPTS + nA];
    rowDesc = pr * 64 + pc;
  }

  // A fragment (ISA 16-bit 16x32 layout): two 16B chunks per k-half.
  union U { u32x4 u[2]; v16bf v; };
  const unsigned short* ar = rows + (size_t)rowDesc * CHN + hi * 8;
  U A0, A1;
  A0.u[0] = *(const u32x4*)(ar);
  A0.u[1] = *(const u32x4*)(ar + 16);
  A1.u[0] = *(const u32x4*)(ar + 32);
  A1.u[1] = *(const u32x4*)(ar + 48);

  // Query grid positions for the 8 accumulator rows this lane owns.
  int qr[8], qc[8];
#pragma unroll
  for (int r = 0; r < 8; ++r) {
    int nr = nbase + r + 8 * hi;
    if (pass == 0) {
      qr[r] = pos2[b * 2 * NPTS + nr];
      qc[r] = pos2[b * 2 * NPTS + NPTS + nr];
    } else {
      int id = ids[b * NPTS + nr];
      qr[r] = id >> 6;
      qc[r] = id & 63;
    }
  }

  float runmin[8];
#pragma unroll
  for (int r = 0; r < 8; ++r) runmin[r] = 1e30f;

  const int bko = hi * 8;       // B fragment: two 8-channel chunks per k-half (table order)
  const int t   = threadIdx.x;

  // Prologue: stage panel 0 (grid row 0) into buffer 0.
  copy_panel(cols, lds_base, 0, 0, t);

  for (int mr = 0; mr < 64; ++mr) {        // panel == key grid row
    wait_async_le0();                      // this wave's slice of panel mr done
    __syncthreads();                       // panel mr visible to all waves;
                                           // buffer (mr+1)&1 no longer being read
    if (mr + 1 < 64)
      copy_panel(cols, lds_base, mr + 1, (mr + 1) & 1, t);  // overlaps compute

    const unsigned short* pan = smem + (mr & 1) * PANEL_US;

    // Burst-load ALL four tiles' B fragments into distinct register sets.
    U B0[4], B1[4];
#pragma unroll
    for (int mt = 0; mt < 4; ++mt) {
      const unsigned short* br = pan + (mt * 16 + lo) * ROW_US + bko;
      B0[mt].u[0] = *(const u32x4*)(br);
      B0[mt].u[1] = *(const u32x4*)(br + 16);
      B1[mt].u[0] = *(const u32x4*)(br + 32);
      B1[mt].u[1] = *(const u32x4*)(br + 48);
    }

    // VALU work inside the load shadow.
    int drok[8];
#pragma unroll
    for (int r = 0; r < 8; ++r) {
      int d = qr[r] - mr; d = d < 0 ? -d : d;
      drok[r] = (d <= 4);
    }

    // Scheduling fence: nothing crosses. All 16 ds_loads stay issued above;
    // WMMAs below get staggered s_wait_dscnt instead of 0x0 each.
    __builtin_amdgcn_sched_barrier(0);

#pragma unroll
    for (int mt = 0; mt < 4; ++mt) {
      v8f acc = {};
      acc = __builtin_amdgcn_wmma_f32_16x16x32_bf16(false, A0.v, false, B0[mt].v,
                                                    (short)0, acc, false, false);
      acc = __builtin_amdgcn_wmma_f32_16x16x32_bf16(false, A1.v, false, B1[mt].v,
                                                    (short)0, acc, false, false);

      int mc = mt * 16 + lo;
#pragma unroll
      for (int r = 0; r < 8; ++r) {
        int dc = qc[r] - mc; dc = dc < 0 ? -dc : dc;
        float pen = (drok[r] && dc <= 4) ? 10.0f : 0.0f;
        float v = fmaf(-2.0f, acc[r], pen);  // (dist - 2) = pen - 2*dot
        runmin[r] = fminf(runmin[r], v);
      }
    }
  }

  // Min across the 16 column-lanes of each half-wave.
#pragma unroll
  for (int off = 1; off < 16; off <<= 1) {
#pragma unroll
    for (int r = 0; r < 8; ++r)
      runmin[r] = fminf(runmin[r], __shfl_xor(runmin[r], off, 32));
  }

  if (lo == 0) {
    float* o = (pass ? neg1 : neg2) + b * NPTS;
#pragma unroll
    for (int r = 0; r < 8; ++r) o[nbase + r + 8 * hi] = 2.f + runmin[r];
  }
}

// ---------------------------------------------------------------------------
// Kernel 4: loss = mean(relu(1 + positive - min(neg1, neg2)))
// ---------------------------------------------------------------------------
__global__ __launch_bounds__(256) void reduce_k(const float* __restrict__ posv,
                                                const float* __restrict__ neg1,
                                                const float* __restrict__ neg2,
                                                float* __restrict__ out) {
  __shared__ float sm[256];
  int tid = threadIdx.x;
  float a = 0.f;
  for (int i = tid; i < BATCH * NPTS; i += 256) {
    float ng = fminf(neg1[i], neg2[i]);
    float d  = 1.0f + posv[i] - ng;
    a += fmaxf(d, 0.0f);
  }
  sm[tid] = a;
  __syncthreads();
  for (int s = 128; s > 0; s >>= 1) {
    if (tid < s) sm[tid] += sm[tid + s];
    __syncthreads();
  }
  if (tid == 0) out[0] = sm[0] / (float)(BATCH * NPTS);
}

// ---------------------------------------------------------------------------
extern "C" void kernel_launch(void* const* d_in, const int* in_sizes, int n_in,
                              void* d_out, int out_size, void* d_ws, size_t ws_size,
                              hipStream_t stream) {
  (void)in_sizes; (void)n_in; (void)out_size; (void)ws_size;
  const float* x1  = (const float*)d_in[0];
  const float* x2  = (const float*)d_in[1];
  const int*   ids = (const int*)d_in[2];
  const int*   p2  = (const int*)d_in[3];

  char* w = (char*)d_ws;
  unsigned short* dT1 = (unsigned short*)w;                       // 4 MB
  unsigned short* dT2 = dT1 + (size_t)BATCH * HW * CHN;           // 4 MB
  float* posv = (float*)(w + (size_t)2 * BATCH * HW * CHN * 2);   // 64 KB
  float* neg1 = posv + BATCH * NPTS;                              // 64 KB
  float* neg2 = neg1 + BATCH * NPTS;                              // 64 KB

  normalize_k<<<dim3(256), dim3(256), 0, stream>>>(x1, x2, dT1, dT2);
  positive_k <<<dim3(64),  dim3(256), 0, stream>>>(dT1, dT2, ids, p2, posv);
  negmin_k   <<<dim3(16, 8, 2), dim3(256), 0, stream>>>(dT1, dT2, ids, p2, neg1, neg2);
  reduce_k   <<<dim3(1),   dim3(256), 0, stream>>>(posv, neg1, neg2, (float*)d_out);
}
